// HGDM_18502719111840
// MI455X (gfx1250) — compile-verified, matrix operand order checked by reading, not run on hardware
//
#include <hip/hip_runtime.h>
#include <hip/hip_bf16.h>
#include <math.h>

#define N_DRUG 3072
#define N_ALL  6144
#define D_F    64
#define EMB    16
#define HEADS  4
#define HD     16

typedef _Float16 half16 __attribute__((ext_vector_type(16)));
typedef float    floatx8 __attribute__((ext_vector_type(8)));
typedef float    floatx4 __attribute__((ext_vector_type(4)));
typedef unsigned int uint32x4 __attribute__((ext_vector_type(4)));
typedef int          int32x8  __attribute__((ext_vector_type(8)));
typedef int          int32x4  __attribute__((ext_vector_type(4)));

__device__ __forceinline__ half16 h16zero() {
    half16 z;
#pragma unroll
    for (int i = 0; i < 16; ++i) z[i] = (_Float16)0.f;
    return z;
}
__device__ __forceinline__ floatx8 f8zero() {
    floatx8 z;
#pragma unroll
    for (int i = 0; i < 8; ++i) z[i] = 0.f;
    return z;
}

__device__ __forceinline__ floatx8 wmma_f16(half16 a, half16 b, floatx8 c) {
    // D = A(16x32 f16) * B(32x16 f16) + C(16x16 f32)
    return __builtin_amdgcn_wmma_f32_16x16x32_f16(false, a, false, b, (short)0, c,
                                                  false, false);
}

// 16 contiguous f32 -> half16 (16B alignment; works for global or LDS pointers)
__device__ __forceinline__ half16 load_cvt16(const float* __restrict__ p) {
    const floatx4* p4 = (const floatx4*)p;
    floatx4 f0 = p4[0], f1 = p4[1], f2 = p4[2], f3 = p4[3];
    half16 r;
#pragma unroll
    for (int i = 0; i < 4; ++i) {
        r[i]      = (_Float16)f0[i];
        r[4 + i]  = (_Float16)f1[i];
        r[8 + i]  = (_Float16)f2[i];
        r[12 + i] = (_Float16)f3[i];
    }
    return r;
}
__device__ __forceinline__ half16 load_h16(const _Float16* __restrict__ p) {
    return *(const half16*)p;  // 32B aligned by construction
}

// ---- Tensor Data Mover: DMA a 64x64 f32 tile (row stride 6144 elems) into LDS.
// D# built per CDNA5 ISA ch.8: group0 = {count/lds_addr/global_addr/type},
// group1 = {data_size, tensor dims, tile dims, strides}. 2D tile -> groups 2/3 zero.
// This toolchain exposes the 6-arg builtin (g0, g1, g2, g3, g4, cpol).
__device__ __forceinline__ void tdm_load_tile64x64(const float* gaddr, unsigned lds_off) {
    unsigned long long ga = (unsigned long long)(uintptr_t)gaddr;
    uint32x4 g0;
    g0[0] = 1u;                                            // count=1 (valid), user mode
    g0[1] = lds_off;                                       // lds_addr (bytes)
    g0[2] = (unsigned)(ga & 0xffffffffu);                  // global_addr[31:0]
    g0[3] = (unsigned)((ga >> 32) & 0x01ffffffu) | (2u << 30);  // addr[56:32] | type=2
    int32x8 g1;
    g1[0] = 0x00020000;                 // workgroup_mask=0, data_size=2 (4 bytes)
    g1[1] = (int)(6144u << 16);         // [47:32]=0, tensor_dim0[15:0]=6144 in [31:16]
    g1[2] = (int)(6144u << 16);         // tensor_dim0[31:16]=0, tensor_dim1[15:0]=6144
    g1[3] = (int)(64u << 16);           // tensor_dim1[31:16]=0, tile_dim0=64
    g1[4] = 64;                         // tile_dim1=64, tile_dim2=0
    g1[5] = 6144;                       // tensor_dim0_stride[31:0] (elements)
    g1[6] = 0;                          // stride hi / dim1_stride lo
    g1[7] = 0;
    int32x4 gz = {0, 0, 0, 0};
    int32x8 gz8 = {0, 0, 0, 0, 0, 0, 0, 0};
    __builtin_amdgcn_tensor_load_to_lds(g0, g1, gz, gz, gz8, 0);
}

// ---------------------------------------------------------------- row norms
__global__ void k_rownorm(const float* __restrict__ graph, float* __restrict__ norm) {
    const int row = blockIdx.x;
    const float* g = graph + (size_t)row * N_ALL;
    float s = 0.f;
    for (int j = threadIdx.x; j < N_ALL; j += 256) s += g[j];
    __shared__ float red[256];
    red[threadIdx.x] = s;
    __syncthreads();
    for (int off = 128; off > 0; off >>= 1) {
        if (threadIdx.x < off) red[threadIdx.x] += red[threadIdx.x + off];
        __syncthreads();
    }
    if (threadIdx.x == 0) {
        float d = red[0];
        if (d < 1.f) d = 1.f;
        norm[row] = rsqrtf(d);
    }
}

// --------------------------------- per-type projection, scale by norm, transpose
// xsT[d][node] = (feat[node] @ W_type)[d] * norm[node]   (f16)
__global__ void k_proj(const float* __restrict__ drug_f, const float* __restrict__ disease_f,
                       const float* __restrict__ drug_w, const float* __restrict__ disease_w,
                       const float* __restrict__ norm, _Float16* __restrict__ xsT) {
    const int row0 = blockIdx.x * 16;
    const bool isDrug = row0 < N_DRUG;
    const float* F = isDrug ? (drug_f + (size_t)row0 * D_F)
                            : (disease_f + (size_t)(row0 - N_DRUG) * D_F);
    const float* W = isDrug ? drug_w : disease_w;
    const int lane = threadIdx.x & 31;
    const int mr = lane & 15, hi = lane >> 4, koff = hi * 16;

    half16 a0 = load_cvt16(F + mr * D_F + koff);
    half16 a1 = load_cvt16(F + mr * D_F + 32 + koff);

    floatx8 acc[4];
#pragma unroll
    for (int ct = 0; ct < 4; ++ct) {
        const int col = ct * 16 + mr;
        half16 b0, b1;
#pragma unroll
        for (int i = 0; i < 16; ++i) {
            b0[i] = (_Float16)W[(koff + i) * D_F + col];
            b1[i] = (_Float16)W[(32 + koff + i) * D_F + col];
        }
        floatx8 c = f8zero();
        c = wmma_f16(a0, b0, c);
        c = wmma_f16(a1, b1, c);
        acc[ct] = c;
    }
#pragma unroll
    for (int ct = 0; ct < 4; ++ct)
#pragma unroll
        for (int v = 0; v < 8; ++v) {
            const int node = row0 + v + 8 * hi;
            const int d = ct * 16 + mr;
            xsT[(size_t)d * N_ALL + node] = (_Float16)(acc[ct][v] * norm[node]);
        }
}

// ------------------------------------- target = (graph @ xs) * norm  (big GEMM)
// 128 threads = 4 waves. TDM double-buffers 64x64 f32 graph tiles into LDS;
// wave 0 issues the descriptors and gates readiness with s_wait_tensorcnt.
__global__ void k_prop(const float* __restrict__ graph, const _Float16* __restrict__ xsT,
                       const float* __restrict__ norm, float* __restrict__ target) {
    const int wave = threadIdx.x >> 5;
    const int lane = threadIdx.x & 31;
    const int mr = lane & 15, hi = lane >> 4, koff = hi * 16;
    const int rowg = blockIdx.x * 64;            // block's 64-row stripe
    const int row0 = rowg + wave * 16;           // this wave's 16 rows

    __shared__ __align__(16) float atile[2][64][64];   // 32KB double buffer
    const bool leader = (threadIdx.x >> 5) == 0;

    floatx8 acc[4];
#pragma unroll
    for (int ct = 0; ct < 4; ++ct) acc[ct] = f8zero();

    if (leader)
        tdm_load_tile64x64(graph + (size_t)rowg * N_ALL,
                           (unsigned)(uintptr_t)&atile[0][0][0]);

    int buf = 0;
    for (int kb = 0; kb < N_ALL; kb += 64, buf ^= 1) {
        if (leader) {
            if (kb + 64 < N_ALL) {
                tdm_load_tile64x64(graph + (size_t)rowg * N_ALL + kb + 64,
                                   (unsigned)(uintptr_t)&atile[buf ^ 1][0][0]);
                __builtin_amdgcn_s_wait_tensorcnt(1);   // current buffer landed
            } else {
                __builtin_amdgcn_s_wait_tensorcnt(0);
            }
        }
        __syncthreads();                                 // tile visible to all waves

        __builtin_prefetch(xsT + (size_t)mr * N_ALL + kb + 512, 0, 1);
#pragma unroll
        for (int s = 0; s < 2; ++s) {                    // two K=32 steps per tile
            half16 a = load_cvt16(&atile[buf][wave * 16 + mr][s * 32 + koff]);
#pragma unroll
            for (int ct = 0; ct < 4; ++ct) {
                half16 b = load_h16(xsT + (size_t)(ct * 16 + mr) * N_ALL +
                                    kb + s * 32 + koff);
                acc[ct] = wmma_f16(a, b, acc[ct]);
            }
        }
        __syncthreads();                                 // before buffer reuse
    }

#pragma unroll
    for (int ct = 0; ct < 4; ++ct)
#pragma unroll
        for (int v = 0; v < 8; ++v) {
            const int r = row0 + v + 8 * hi;
            target[(size_t)r * D_F + ct * 16 + mr] = acc[ct][v] * norm[r];
        }
}

// ----------------------------------------- denoise MLP (80 -> 256 -> 64), per node
__global__ void k_mlp(const float* __restrict__ target, const int* __restrict__ timesteps,
                      const float* __restrict__ emb_w, const float* __restrict__ emb_b,
                      const float* __restrict__ in_w, const float* __restrict__ in_b,
                      const float* __restrict__ out_w, const float* __restrict__ out_b,
                      float* __restrict__ aux) {
    const int node = blockIdx.x;
    const int t = threadIdx.x;  // 256
    __shared__ float xin[80];
    __shared__ float h[256];
    if (t < 64) {
        xin[t] = target[(size_t)node * D_F + t];
    } else if (t < 80) {
        const int j = t - 64;
        const float ts = (float)timesteps[node];
        float e = 0.f;
#pragma unroll
        for (int ee = 0; ee < 16; ++ee) {
            const float fr = __expf(-9.210340371976184f * (float)(ee & 7) * 0.125f);
            const float te = (ee < 8) ? __cosf(ts * fr) : __sinf(ts * fr);
            e += te * emb_w[ee * EMB + j];
        }
        xin[t] = e + emb_b[j];
    }
    __syncthreads();
    float hv = in_b[t];
    for (int k = 0; k < 80; ++k) hv += xin[k] * in_w[k * 256 + t];
    h[t] = tanhf(hv);
    __syncthreads();
    if (t < 64) {
        float a = out_b[t];
        for (int k = 0; k < 256; ++k) a += h[k] * out_w[k * D_F + t];
        aux[(size_t)node * D_F + t] = a;
    }
}

// -------------------------------------------- QKV projections (WMMA, f16 outputs)
__device__ __forceinline__ void proj64(half16 a0, half16 a1, const float* __restrict__ W,
                                       int mr, int koff, floatx8 out[4]) {
#pragma unroll
    for (int ct = 0; ct < 4; ++ct) {
        const int col = ct * 16 + mr;
        half16 b0, b1;
#pragma unroll
        for (int i = 0; i < 16; ++i) {
            b0[i] = (_Float16)W[(koff + i) * D_F + col];
            b1[i] = (_Float16)W[(32 + koff + i) * D_F + col];
        }
        floatx8 c = f8zero();
        c = wmma_f16(a0, b0, c);
        c = wmma_f16(a1, b1, c);
        out[ct] = c;
    }
}

__global__ void k_qkv(const float* __restrict__ target, const float* __restrict__ aux,
                      const float* __restrict__ wq, const float* __restrict__ bq,
                      const float* __restrict__ wk, const float* __restrict__ bk,
                      const float* __restrict__ wv, const float* __restrict__ bv,
                      _Float16* __restrict__ qh, _Float16* __restrict__ kh,
                      _Float16* __restrict__ vT) {
    const int row0 = blockIdx.x * 16;
    const int lane = threadIdx.x & 31;
    const int mr = lane & 15, hi = lane >> 4, koff = hi * 16;

    half16 at0 = load_cvt16(target + (size_t)(row0 + mr) * D_F + koff);
    half16 at1 = load_cvt16(target + (size_t)(row0 + mr) * D_F + 32 + koff);
    half16 aa0 = load_cvt16(aux + (size_t)(row0 + mr) * D_F + koff);
    half16 aa1 = load_cvt16(aux + (size_t)(row0 + mr) * D_F + 32 + koff);

    floatx8 acc[4];
    // Q (scale 1/sqrt(hd)=0.25 folded in)
    proj64(at0, at1, wq, mr, koff, acc);
#pragma unroll
    for (int ct = 0; ct < 4; ++ct)
#pragma unroll
        for (int v = 0; v < 8; ++v) {
            const int node = row0 + v + 8 * hi, d = ct * 16 + mr;
            qh[(size_t)node * D_F + d] = (_Float16)((acc[ct][v] + bq[d]) * 0.25f);
        }
    // K
    proj64(aa0, aa1, wk, mr, koff, acc);
#pragma unroll
    for (int ct = 0; ct < 4; ++ct)
#pragma unroll
        for (int v = 0; v < 8; ++v) {
            const int node = row0 + v + 8 * hi, d = ct * 16 + mr;
            kh[(size_t)node * D_F + d] = (_Float16)(acc[ct][v] + bk[d]);
        }
    // V (transposed store for B-fragment-friendly P·V)
    proj64(aa0, aa1, wv, mr, koff, acc);
#pragma unroll
    for (int ct = 0; ct < 4; ++ct)
#pragma unroll
        for (int v = 0; v < 8; ++v) {
            const int node = row0 + v + 8 * hi, d = ct * 16 + mr;
            vT[(size_t)d * N_ALL + node] = (_Float16)(acc[ct][v] + bv[d]);
        }
}

// --------------------------- flash-style attention: 16 queries x 1 head per block
// 8 waves split the 6144 keys (768 each). Two passes: row-max, then exp+sum+P*V.
__global__ void k_attn(const _Float16* __restrict__ qh, const _Float16* __restrict__ kh,
                       const _Float16* __restrict__ vT, float* __restrict__ attn_pre) {
    const int head = blockIdx.y;
    const int q0 = blockIdx.x * 16;
    const int wave = threadIdx.x >> 5;
    const int lane = threadIdx.x & 31;
    const int mr = lane & 15, hi = lane >> 4;
    const int kc0 = wave * 768;

    __shared__ __align__(32) _Float16 pstage[8][16 * 32];  // per-wave P tile
    __shared__ float wred[8][16];
    __shared__ float rowmax[16];
    __shared__ float rowsum[16];
    __shared__ float opart[8][16][16];

    // Q A-fragment: real head dims in lower lane-half, zero padding (K=16..31) above
    half16 aq = h16zero();
    if (hi == 0) aq = load_h16(qh + (size_t)(q0 + mr) * D_F + head * HD);

    // ---------------- pass 1: row maxima over this wave's keys
    float m[8];
#pragma unroll
    for (int v = 0; v < 8; ++v) m[v] = -1e30f;
    for (int kt = kc0; kt < kc0 + 768; kt += 16) {
        half16 bkf = h16zero();
        if (hi == 0) bkf = load_h16(kh + (size_t)(kt + mr) * D_F + head * HD);
        floatx8 s = wmma_f16(aq, bkf, f8zero());
#pragma unroll
        for (int v = 0; v < 8; ++v) m[v] = fmaxf(m[v], s[v]);
    }
#pragma unroll
    for (int off = 1; off < 16; off <<= 1)
#pragma unroll
        for (int v = 0; v < 8; ++v) m[v] = fmaxf(m[v], __shfl_xor(m[v], off, 32));
    if (mr == 0)
#pragma unroll
        for (int v = 0; v < 8; ++v) wred[wave][8 * hi + v] = m[v];
    __syncthreads();
    if (threadIdx.x < 16) {
        float r = -1e30f;
        for (int w = 0; w < 8; ++w) r = fmaxf(r, wred[w][threadIdx.x]);
        rowmax[threadIdx.x] = r;
    }
    __syncthreads();

    // ---------------- pass 2: exp, row sums, O = P * V
    floatx8 oacc = f8zero();
    float sm[8];
#pragma unroll
    for (int v = 0; v < 8; ++v) sm[v] = 0.f;

    for (int kt2 = kc0; kt2 < kc0 + 768; kt2 += 32) {
#pragma unroll
        for (int sub = 0; sub < 2; ++sub) {
            const int kt = kt2 + sub * 16;
            half16 bkf = h16zero();
            if (hi == 0) bkf = load_h16(kh + (size_t)(kt + mr) * D_F + head * HD);
            floatx8 s = wmma_f16(aq, bkf, f8zero());
#pragma unroll
            for (int v = 0; v < 8; ++v) {
                const int r = v + 8 * hi;
                const float p = __expf(s[v] - rowmax[r]);
                sm[v] += p;
                pstage[wave][r * 32 + sub * 16 + mr] = (_Float16)p;
            }
        }
        __syncthreads();  // cross-lane LDS visibility (uniform trip count)
        half16 ap = load_h16(&pstage[wave][mr * 32 + hi * 16]);
        half16 bvf = load_h16(vT + (size_t)(head * HD + mr) * N_ALL + kt2 + hi * 16);
        oacc = wmma_f16(ap, bvf, oacc);
        __syncthreads();
    }

#pragma unroll
    for (int off = 1; off < 16; off <<= 1)
#pragma unroll
        for (int v = 0; v < 8; ++v) sm[v] += __shfl_xor(sm[v], off, 32);
    if (mr == 0)
#pragma unroll
        for (int v = 0; v < 8; ++v) wred[wave][8 * hi + v] = sm[v];
#pragma unroll
    for (int v = 0; v < 8; ++v) opart[wave][v + 8 * hi][mr] = oacc[v];
    __syncthreads();
    if (threadIdx.x < 16) {
        float r = 0.f;
        for (int w = 0; w < 8; ++w) r += wred[w][threadIdx.x];
        rowsum[threadIdx.x] = r;
    }
    __syncthreads();
    {
        const int q = threadIdx.x >> 4, d = threadIdx.x & 15;
        float val = 0.f;
        for (int w = 0; w < 8; ++w) val += opart[w][q][d];
        attn_pre[(size_t)(q0 + q) * D_F + head * HD + d] = val / rowsum[q];
    }
}

// ----------------------- out-proj + fuse + residual LayerNorm, per node (64 thr)
__global__ void k_out(const float* __restrict__ target, const float* __restrict__ attn_pre,
                      const float* __restrict__ wo, const float* __restrict__ bo,
                      const float* __restrict__ fuse_w, const float* __restrict__ fuse_b,
                      const float* __restrict__ ln_g, const float* __restrict__ ln_b,
                      float* __restrict__ out) {
    const int node = blockIdx.x;
    const int d = threadIdx.x;  // 64
    __shared__ float tg[64], ap[64], ao[64], part[2];

    tg[d] = target[(size_t)node * D_F + d];
    ap[d] = attn_pre[(size_t)node * D_F + d];
    __syncthreads();

    float a = bo[d];
    for (int k = 0; k < 64; ++k) a += ap[k] * wo[k * D_F + d];
    ao[d] = a;
    __syncthreads();

    float f = fuse_b[d];
    for (int k = 0; k < 64; ++k) f += tg[k] * fuse_w[k * D_F + d];
    for (int k = 0; k < 64; ++k) f += ao[k] * fuse_w[(64 + k) * D_F + d];
    const float r = f + tg[d];

    float s = r;
#pragma unroll
    for (int off = 16; off > 0; off >>= 1) s += __shfl_xor(s, off, 32);
    if ((d & 31) == 0) part[d >> 5] = s;
    __syncthreads();
    const float mu = (part[0] + part[1]) * (1.f / 64.f);
    __syncthreads();

    const float dv = r - mu;
    float s2 = dv * dv;
#pragma unroll
    for (int off = 16; off > 0; off >>= 1) s2 += __shfl_xor(s2, off, 32);
    if ((d & 31) == 0) part[d >> 5] = s2;
    __syncthreads();
    const float var = (part[0] + part[1]) * (1.f / 64.f);

    out[(size_t)node * D_F + d] = dv * rsqrtf(var + 1e-5f) * ln_g[d] + ln_b[d];
}

// ------------------------------------------------------------------- launcher
extern "C" void kernel_launch(void* const* d_in, const int* in_sizes, int n_in,
                              void* d_out, int out_size, void* d_ws, size_t ws_size,
                              hipStream_t stream) {
    const float* graph     = (const float*)d_in[0];
    const float* drug_f    = (const float*)d_in[1];
    const float* disease_f = (const float*)d_in[2];
    const int*   timesteps = (const int*)d_in[3];
    const float* drug_w    = (const float*)d_in[4];
    const float* disease_w = (const float*)d_in[5];
    const float* emb_w     = (const float*)d_in[6];
    const float* emb_b     = (const float*)d_in[7];
    const float* in_w      = (const float*)d_in[8];
    const float* in_b      = (const float*)d_in[9];
    const float* out_w     = (const float*)d_in[10];
    const float* out_b     = (const float*)d_in[11];
    const float* wq        = (const float*)d_in[12];
    const float* bq        = (const float*)d_in[13];
    const float* wk        = (const float*)d_in[14];
    const float* bk        = (const float*)d_in[15];
    const float* wv        = (const float*)d_in[16];
    const float* bv        = (const float*)d_in[17];
    const float* wo        = (const float*)d_in[18];
    const float* bo        = (const float*)d_in[19];
    const float* fuse_w    = (const float*)d_in[20];
    const float* fuse_b    = (const float*)d_in[21];
    const float* ln_g      = (const float*)d_in[22];
    const float* ln_b      = (const float*)d_in[23];
    float* out = (float*)d_out;

    char* ws = (char*)d_ws;
    float*    norm     = (float*)(ws);                              // 24576 B
    _Float16* xsT      = (_Float16*)(ws + 24576);                   // 786432 B
    float*    target   = (float*)(ws + 811008);                     // 1572864 B
    float*    aux      = (float*)(ws + 2383872);                    // 1572864 B
    _Float16* qh       = (_Float16*)(ws + 3956736);                 // 786432 B
    _Float16* kh       = (_Float16*)(ws + 4743168);                 // 786432 B
    _Float16* vT       = (_Float16*)(ws + 5529600);                 // 786432 B
    float*    attn_pre = (float*)(ws + 6316032);                    // 1572864 B

    k_rownorm<<<N_ALL, 256, 0, stream>>>(graph, norm);
    k_proj<<<N_ALL / 16, 32, 0, stream>>>(drug_f, disease_f, drug_w, disease_w, norm, xsT);
    k_prop<<<N_ALL / 64, 128, 0, stream>>>(graph, xsT, norm, target);
    k_mlp<<<N_ALL, 256, 0, stream>>>(target, timesteps, emb_w, emb_b, in_w, in_b,
                                     out_w, out_b, aux);
    k_qkv<<<N_ALL / 16, 32, 0, stream>>>(target, aux, wq, bq, wk, bk, wv, bv, qh, kh, vT);
    k_attn<<<dim3(N_ALL / 16, HEADS), 256, 0, stream>>>(qh, kh, vT, attn_pre);
    k_out<<<N_ALL, 64, 0, stream>>>(target, attn_pre, wo, bo, fuse_w, fuse_b,
                                    ln_g, ln_b, out);
}
